// RecurrentGCN_44160853737700
// MI455X (gfx1250) — matrix operand order repaired and checked by reading, not compiled, not run on hardware
//
#include <hip/hip_runtime.h>
#include <hip/hip_bf16.h>
#include <math.h>

typedef __attribute__((ext_vector_type(16))) __bf16 v16bf;
typedef __attribute__((ext_vector_type(8)))  float  v8f;

#define F_IN 128
#define HID  32

// Fused DCRNN(K=1) cell + linear readout.
// out[n] = relu((1-sigmoid(x@Wz+bz)) * tanh(x@Wh+bh)) @ W_lin + b_lin
// Wz/Wh effective = W[0,0,:128,:] + W[1,0,:128,:]  (H0 == 0 kills the rest)
__global__ __launch_bounds__(256) void dcrnn_fused_wmma(
    const float* __restrict__ x,     // [N,128]
    const float* __restrict__ Wz,    // (2,1,160,32) flat
    const float* __restrict__ bz,    // [32]
    const float* __restrict__ Wh,    // (2,1,160,32) flat
    const float* __restrict__ bh,    // [32]
    const float* __restrict__ wlin,  // [32]
    const float* __restrict__ blin,  // [1]
    float* __restrict__ out,         // [N]
    int N)
{
    // 16 B-fragments (2 gates x 4 k-steps x 2 n-tiles), each 32 lanes x 16 bf16.
    __shared__ __align__(32) __bf16 sW[16 * 512];   // 16 KB

    const int tid = threadIdx.x;

    // ---- stage combined weights into WMMA B-matrix lane layout (bf16) ----
    // B 32x16 bf16, this code's k order: lane = N + (K>=16 ? 16 : 0) half; element e -> K = Kbase + e.
    for (int q = tid; q < 512; q += 256) {
        const int frag = q >> 5;          // 0..15
        const int lane = q & 31;
        const int g    = frag >> 3;       // 0 = z, 1 = h
        const int s    = (frag >> 1) & 3; // k-step (32 K per step)
        const int t    = frag & 1;        // n-tile (16 cols per tile)
        const float* W = g ? Wh : Wz;
        const int n  = t * 16 + (lane & 15);
        const int kb = 32 * s + ((lane & 16) ? 16 : 0);
        __bf16* dst = &sW[frag * 512 + lane * 16];
        #pragma unroll
        for (int e = 0; e < 16; ++e) {
            const int k = kb + e;                          // k < 128
            const float v = W[k * 32 + n] + W[5120 + k * 32 + n]; // fwd + bwd term
            dst[e] = (__bf16)v;
        }
    }
    __syncthreads();

    const int lane = tid & 31;
    const int wave = tid >> 5;
    const int tb   = (blockIdx.x * 8 + wave) * 16;  // tile base row
    if (tb >= N) return;                            // uniform per-wave: EXEC stays full

    v8f az0 = {}; v8f az1 = {}; v8f ah0 = {}; v8f ah1 = {};

    // A 16x32 bf16, this code's k order (matches B): lane m = lane&15; Kbase = (lane<16 ? 0 : 16);
    // elems 0..15 -> K = Kbase..Kbase+15.
    const int m  = lane & 15;
    int row = tb + m; if (row >= N) row = N - 1;
    const float* xr = x + (long)row * F_IN;
    const int kb = (lane & 16) ? 16 : 0;

    #pragma unroll
    for (int s = 0; s < 4; ++s) {
        const float4* p0 = (const float4*)(xr + 32 * s + kb);
        const float4  f0 = p0[0];
        const float4  f1 = p0[1];
        const float4* p1 = (const float4*)(xr + 32 * s + kb + 8);
        const float4  f2 = p1[0];
        const float4  f3 = p1[1];
        v16bf a;
        a[0]=(__bf16)f0.x;  a[1]=(__bf16)f0.y;  a[2]=(__bf16)f0.z;  a[3]=(__bf16)f0.w;
        a[4]=(__bf16)f1.x;  a[5]=(__bf16)f1.y;  a[6]=(__bf16)f1.z;  a[7]=(__bf16)f1.w;
        a[8]=(__bf16)f2.x;  a[9]=(__bf16)f2.y;  a[10]=(__bf16)f2.z; a[11]=(__bf16)f2.w;
        a[12]=(__bf16)f3.x; a[13]=(__bf16)f3.y; a[14]=(__bf16)f3.z; a[15]=(__bf16)f3.w;

        const v16bf bz0f = *(const v16bf*)&sW[((0*4 + s)*2 + 0) * 512 + lane * 16];
        const v16bf bz1f = *(const v16bf*)&sW[((0*4 + s)*2 + 1) * 512 + lane * 16];
        const v16bf bh0f = *(const v16bf*)&sW[((1*4 + s)*2 + 0) * 512 + lane * 16];
        const v16bf bh1f = *(const v16bf*)&sW[((1*4 + s)*2 + 1) * 512 + lane * 16];

        az0 = __builtin_amdgcn_wmma_f32_16x16x32_bf16(false, a, false, bz0f, (short)0, az0, false, false);
        az1 = __builtin_amdgcn_wmma_f32_16x16x32_bf16(false, a, false, bz1f, (short)0, az1, false, false);
        ah0 = __builtin_amdgcn_wmma_f32_16x16x32_bf16(false, a, false, bh0f, (short)0, ah0, false, false);
        ah1 = __builtin_amdgcn_wmma_f32_16x16x32_bf16(false, a, false, bh1f, (short)0, ah1, false, false);
    }

    // ---- epilogue: gates + readout ----
    // C/D layout: VGPR v -> row M = v + 8*(lane>=16), col N = lane&15 (+16 for tile1)
    const int   n0  = lane & 15;
    const int   n1  = n0 + 16;
    const float bzv0 = bz[n0],  bzv1 = bz[n1];
    const float bhv0 = bh[n0],  bhv1 = bh[n1];
    const float wl0  = wlin[n0], wl1 = wlin[n1];
    const float bl   = blin[0];

    float r[8];
    #pragma unroll
    for (int v = 0; v < 8; ++v) {
        const float z0 = 1.0f / (1.0f + __expf(-(az0[v] + bzv0)));
        const float z1 = 1.0f / (1.0f + __expf(-(az1[v] + bzv1)));
        const float t0 = tanhf(ah0[v] + bhv0);
        const float t1 = tanhf(ah1[v] + bhv1);
        const float h0 = fmaxf((1.0f - z0) * t0, 0.0f);
        const float h1 = fmaxf((1.0f - z1) * t1, 0.0f);
        r[v] = h0 * wl0 + h1 * wl1;
    }
    // butterfly reduce across the 16 lanes of each half (covers all 32 HID cols)
    #pragma unroll
    for (int off = 1; off < 16; off <<= 1) {
        #pragma unroll
        for (int v = 0; v < 8; ++v) r[v] += __shfl_xor(r[v], off, 32);
    }
    if ((lane & 15) == 0) {
        const int mb = tb + ((lane >> 4) << 3);
        #pragma unroll
        for (int v = 0; v < 8; ++v) {
            const int rr = mb + v;
            if (rr < N) out[rr] = r[v] + bl;
        }
    }
}

extern "C" void kernel_launch(void* const* d_in, const int* in_sizes, int n_in,
                              void* d_out, int out_size, void* d_ws, size_t ws_size,
                              hipStream_t stream) {
    // setup_inputs order: 0 x, 1 edge_index (dead), 2 edge_weight (dead),
    // 3 W_z, 4 b_z, 5 W_r (dead), 6 b_r (dead), 7 W_h, 8 b_h, 9 W_lin, 10 b_lin
    const float* x    = (const float*)d_in[0];
    const float* Wz   = (const float*)d_in[3];
    const float* bz   = (const float*)d_in[4];
    const float* Wh   = (const float*)d_in[7];
    const float* bh   = (const float*)d_in[8];
    const float* wlin = (const float*)d_in[9];
    const float* blin = (const float*)d_in[10];
    float* out = (float*)d_out;

    const int N      = in_sizes[0] / F_IN;       // 100000
    const int tiles  = (N + 15) / 16;            // 6250
    const int blocks = (tiles + 7) / 8;          // 782

    dcrnn_fused_wmma<<<blocks, 256, 0, stream>>>(x, Wz, bz, Wh, bh, wlin, blin, out, N);
}
